// KuramotoBlock_59949153517854
// MI455X (gfx1250) — matrix operand order repaired and checked by reading, not compiled, not run on hardware
//
#include <hip/hip_runtime.h>
#include <hip/hip_bf16.h>

typedef __bf16 bf16_t;
typedef bf16_t v16bf __attribute__((ext_vector_type(16)));
typedef float  v8f   __attribute__((ext_vector_type(8)));

union Frag {
    uint4 q[2];
    v16bf v;
};

#define HWPIX 9216       // 96*96
#define NPIX  73728      // 8*96*96
#define CCH   128
#define KTOT  6272       // 128*49
#define NK64  98         // KTOT/64
#define ASTRIDE 72       // 64 + 8 bf16 pad (16B-aligned rows)

static __device__ __forceinline__ unsigned short f2bf(float f) {
    unsigned int u = __float_as_uint(f);
    unsigned int r = (u + 0x7FFFu + ((u >> 16) & 1u)) >> 16;
    return (unsigned short)r;
}

// ---------------- init energies (d_out tail is poisoned) ----------------
__global__ void init_energy(float* e_out) {
    int i = threadIdx.x;
    if (i < 40) e_out[i] = 0.0f;
}

// -------- reorder conv_w (O,I,7,7) f32 -> WMMA B-fragment bf16 ----------
// layout: frag = kc32*8 + nt ; within frag: lane*16 + e (contiguous per lane)
// element (lane,e): K = (lane/16)*8 + (e%8) + (e/8)*16 ; N = nt*16 + (lane%16)
__global__ void prep_weights(const float* __restrict__ w, unsigned short* __restrict__ wf) {
    int idx = blockIdx.x * 256 + threadIdx.x;          // [0, 196*8*512)
    if (idx >= 196 * 8 * 512) return;
    int e    = idx & 15;
    int lane = (idx >> 4) & 31;
    int frag = idx >> 9;
    int nt   = frag & 7;
    int kc32 = frag >> 3;
    int K    = ((lane >> 4) << 3) + (e & 7) + ((e >> 3) << 4);
    int n    = nt * 16 + (lane & 15);
    int kg   = kc32 * 32 + K;                          // global K, ordered (ky,kx,ci)
    int kyx  = kg >> 7;
    int ci   = kg & 127;
    int ky   = kyx / 7;
    int kx   = kyx - ky * 7;
    float v  = w[(((size_t)n * 128 + ci) * 7 + ky) * 7 + kx];
    wf[idx]  = f2bf(v);
}

// ------------- GroupNorm(c) NCHW f32 -> NHWC f32 (64 groups of 2ch) ------
__global__ void groupnorm_k(const float* __restrict__ c, const float* __restrict__ gw,
                            const float* __restrict__ gb, float* __restrict__ cn) {
    __shared__ float s1[256];
    __shared__ float s2[256];
    int tid = threadIdx.x;
    int b = blockIdx.x >> 6;
    int g = blockIdx.x & 63;
    const float* src = c + ((size_t)b * CCH + 2 * g) * HWPIX;   // 18432 contiguous floats
    float s = 0.f, q = 0.f;
    for (int i = tid; i < 2 * HWPIX; i += 256) { float v = src[i]; s += v; q += v * v; }
    s1[tid] = s; s2[tid] = q;
    __syncthreads();
    for (int st = 128; st > 0; st >>= 1) {
        if (tid < st) { s1[tid] += s1[tid + st]; s2[tid] += s2[tid + st]; }
        __syncthreads();
    }
    float mu = s1[0] * (1.0f / (2 * HWPIX));
    float var = s2[0] * (1.0f / (2 * HWPIX)) - mu * mu;
    float rs = rsqrtf(var + 1e-5f);
    for (int i = tid; i < 2 * HWPIX; i += 256) {
        int chl = i / HWPIX;
        int pix = i - chl * HWPIX;
        int ch  = 2 * g + chl;
        float v = (src[i] - mu) * rs * gw[ch] + gb[ch];
        cn[((size_t)b * HWPIX + pix) * CCH + ch] = v;
    }
}

// ------- initial oscillator normalize: x NCHW f32 -> NHWC f32 + bf16 -----
__global__ void normx_k(const float* __restrict__ x, float* __restrict__ xf,
                        unsigned short* __restrict__ xb) {
    unsigned int gidx = blockIdx.x * 256 + threadIdx.x;   // pixel*64 + pair
    unsigned int pm = gidx >> 6;
    unsigned int p  = gidx & 63;
    unsigned int b  = pm / HWPIX;
    unsigned int pix = pm - b * HWPIX;
    const float* s = x + ((size_t)b * CCH + 2 * p) * HWPIX + pix;
    float v0 = s[0], v1 = s[HWPIX];
    float n = fmaxf(sqrtf(v0 * v0 + v1 * v1), 1e-12f);
    v0 /= n; v1 /= n;
    int o = pm * CCH + 2 * p;
    xf[o] = v0; xf[o + 1] = v1;
    xb[o] = f2bf(v0); xb[o + 1] = f2bf(v1);
}

// --------------- implicit-GEMM 7x7 conv via WMMA bf16 -------------------
// grid.x = 576 blocks of 128 pixels; block = 256 threads = 8 waves (4M x 2N)
// Staging uses GLOBAL_LOAD_ASYNC_TO_LDS_B128 (ASYNCcnt) for valid taps,
// ds_store of zeros for halo taps; double-buffered LDS, one barrier/chunk.
__global__ __launch_bounds__(256) void conv_wmma(
    const unsigned short* __restrict__ xb,   // NHWC bf16 activations
    const unsigned short* __restrict__ wf,   // pre-swizzled B fragments
    float* __restrict__ coup)                // NHWC f32 coupling out
{
    extern __shared__ char smem_raw[];
    unsigned short* As = (unsigned short*)smem_raw;                           // [2][128][ASTRIDE]
    unsigned short* Bs = (unsigned short*)(smem_raw + 2 * 128 * ASTRIDE * 2); // [2][8192]

    const int tid  = threadIdx.x;
    const int lane = tid & 31;
    const int wv   = tid >> 5;
    const int wm   = wv & 3;   // M quadrant (32 rows each)
    const int wn   = wv >> 2;  // N half (64 cols each)
    const int pmBase = blockIdx.x * 128;

    // ---- per-segment invariants (4 staged 16B segments per thread) ----
    int py[4], pxc[4];
    unsigned long long gA[4], gBb[4];
    unsigned int laA[4], laB[4];
    unsigned short* pA[4];
    #pragma unroll
    for (int i = 0; i < 4; ++i) {
        unsigned int sid = tid + i * 256;        // [0,1024): 128 px * 8 segs
        unsigned int pxl = sid >> 3;
        unsigned int seg = sid & 7;
        unsigned int pm  = pmBase + pxl;
        unsigned int b   = pm / HWPIX;
        unsigned int rem = pm - b * HWPIX;
        unsigned int oy  = rem / 96;
        unsigned int ox  = rem - oy * 96;
        py[i] = oy; pxc[i] = ox;
        gA[i]  = (unsigned long long)(uintptr_t)(xb + ((((b * 96 + oy) * 96 + ox)) << 7) + seg * 8);
        gBb[i] = (unsigned long long)(uintptr_t)(wf + sid * 8);
        pA[i]  = As + pxl * ASTRIDE + seg * 8;
        laA[i] = (unsigned int)(uintptr_t)pA[i];
        laB[i] = (unsigned int)(uintptr_t)(Bs + sid * 8);
    }

    v8f acc[2][4];
    #pragma unroll
    for (int mt = 0; mt < 2; ++mt)
        #pragma unroll
        for (int nt = 0; nt < 4; ++nt)
            acc[mt][nt] = (v8f)(0.0f);

    auto stage = [&](int kc, int buf) {
        unsigned int kyx = (unsigned int)kc >> 1;
        int ky = kyx / 7;
        int kx = kyx - ky * 7;
        long long dA = ((long long)((ky - 3) * 96 + (kx - 3)) * 128 + ((kc & 1) * 64)) * 2;
        unsigned int bufA = (unsigned int)buf * (128 * ASTRIDE * 2);
        unsigned int bufB = (unsigned int)buf * (8192 * 2);
        unsigned long long gOffB = (unsigned long long)kc * 16384ull;
        #pragma unroll
        for (int i = 0; i < 4; ++i) {
            int iy = py[i] + ky - 3;
            int ix = pxc[i] + kx - 3;
            unsigned int la = laA[i] + bufA;
            if ((unsigned int)iy < 96u && (unsigned int)ix < 96u) {
                unsigned long long ga = gA[i] + dA;
                asm volatile("global_load_async_to_lds_b128 %0, %1, off"
                             :: "v"(la), "v"(ga) : "memory");
            } else {
                *(uint4*)(pA[i] + (size_t)buf * (128 * ASTRIDE)) = make_uint4(0u, 0u, 0u, 0u);
            }
            unsigned long long gb = gBb[i] + gOffB;
            unsigned int lb = laB[i] + bufB;
            asm volatile("global_load_async_to_lds_b128 %0, %1, off"
                         :: "v"(lb), "v"(gb) : "memory");
        }
    };

    stage(0, 0);
    asm volatile("s_wait_asynccnt 0x0" ::: "memory");
    __syncthreads();

    #pragma unroll 1
    for (int kc = 0; kc < NK64; ++kc) {
        int buf = kc & 1;
        if (kc + 1 < NK64) stage(kc + 1, buf ^ 1);

        #pragma unroll
        for (int kk = 0; kk < 2; ++kk) {
            Frag a[2];
            #pragma unroll
            for (int mt = 0; mt < 2; ++mt) {
                const unsigned short* ab = As +
                    (buf * 128 + wm * 32 + mt * 16 + (lane & 15)) * ASTRIDE +
                    kk * 32 + (lane >> 4) * 8;
                a[mt].q[0] = *(const uint4*)ab;        // K = kk*32 + h*8 + 0..7
                a[mt].q[1] = *(const uint4*)(ab + 16); // K = kk*32 + 16 + h*8 + 0..7
            }
            #pragma unroll
            for (int nt = 0; nt < 4; ++nt) {
                Frag bf;
                const unsigned short* bb = Bs + buf * 8192 +
                    ((kk * 8 + wn * 4 + nt) << 9) + lane * 16;
                bf.q[0] = *(const uint4*)bb;
                bf.q[1] = *(const uint4*)(bb + 8);
                #pragma unroll
                for (int mt = 0; mt < 2; ++mt) {
                    acc[mt][nt] = __builtin_amdgcn_wmma_f32_16x16x32_bf16(
                        false, a[mt].v, false, bf.v, (short)0, acc[mt][nt], false, false);
                }
            }
        }

        if (kc + 1 < NK64) asm volatile("s_wait_asynccnt 0x0" ::: "memory");
        __syncthreads();
    }

    // epilogue: C/D layout — lane row M=(lane/16)*8+r, col N=lane%16
    #pragma unroll
    for (int mt = 0; mt < 2; ++mt) {
        #pragma unroll
        for (int nt = 0; nt < 4; ++nt) {
            int prow0 = pmBase + wm * 32 + mt * 16 + ((lane >> 4) << 3);
            int ch = wn * 64 + nt * 16 + (lane & 15);
            #pragma unroll
            for (int r = 0; r < 8; ++r) {
                coup[(size_t)(prow0 + r) * CCH + ch] = acc[mt][nt][r];
            }
        }
    }
}

// --------- fused Kuramoto step: drive, project, rotate, renorm ----------
__global__ void kuramoto_k(float* __restrict__ xf, const float* __restrict__ coup,
                           const float* __restrict__ cn, const float* __restrict__ convb,
                           const float* __restrict__ omega_p, unsigned short* __restrict__ xb,
                           float* __restrict__ xs_out, float* __restrict__ e_out, int t) {
    __shared__ float red[256];
    int tid = threadIdx.x;
    int p   = tid & 63;
    int pl  = tid >> 6;
    unsigned int pm  = blockIdx.x * 4 + pl;
    unsigned int b   = pm / HWPIX;          // all 4 pixels in a block share b
    unsigned int pix = pm - b * HWPIX;
    int base = pm * CCH + 2 * p;

    float om = omega_p[0];
    float x0 = xf[base], x1 = xf[base + 1];
    float y0 = coup[base]     + convb[2 * p]     + cn[base];
    float y1 = coup[base + 1] + convb[2 * p + 1] + cn[base + 1];

    float inner = x0 * y0 + x1 * y1;
    float n0 = x0 + (-om * x1) + (y0 - inner * x0);   // gamma = 1
    float n1 = x1 + ( om * x0) + (y1 - inner * x1);
    float nn = fmaxf(sqrtf(n0 * n0 + n1 * n1), 1e-12f);
    n0 /= nn; n1 /= nn;

    xf[base] = n0; xf[base + 1] = n1;
    xb[base] = f2bf(n0); xb[base + 1] = f2bf(n1);

    size_t ob = ((size_t)(t * 8 + b) * CCH + 2 * p) * HWPIX + pix;  // NCHW out
    xs_out[ob] = n0;
    xs_out[ob + HWPIX] = n1;

    red[tid] = -(y0 * x0 + y1 * x1);
    __syncthreads();
    for (int s = 128; s > 0; s >>= 1) {
        if (tid < s) red[tid] += red[tid + s];
        __syncthreads();
    }
    if (tid == 0) atomicAdd(&e_out[(t + 1) * 8 + b], red[0]);
}

extern "C" void kernel_launch(void* const* d_in, const int* in_sizes, int n_in,
                              void* d_out, int out_size, void* d_ws, size_t ws_size,
                              hipStream_t stream) {
    (void)in_sizes; (void)n_in; (void)out_size; (void)ws_size;
    const float* x      = (const float*)d_in[0];
    const float* c      = (const float*)d_in[1];
    const float* conv_w = (const float*)d_in[2];
    const float* conv_b = (const float*)d_in[3];
    const float* gn_w   = (const float*)d_in[4];
    const float* gn_b   = (const float*)d_in[5];
    const float* omega  = (const float*)d_in[6];
    // d_in[7] = n_timesteps (fixed 4 per setup_inputs)

    const size_t TENS = (size_t)NPIX * CCH;            // 9437184 elements
    char* ws = (char*)d_ws;
    float* c_norm          = (float*)(ws);
    float* x_f32           = (float*)(ws + TENS * 4);
    float* coup            = (float*)(ws + TENS * 8);
    unsigned short* x_bf   = (unsigned short*)(ws + TENS * 12);
    unsigned short* w_frag = (unsigned short*)(ws + TENS * 12 + TENS * 2);

    float* xs_out = (float*)d_out;
    float* e_out  = xs_out + 4 * TENS;

    const int SMEM = (2 * 128 * ASTRIDE + 2 * 8192) * 2;  // 69632 bytes

    init_energy<<<1, 64, 0, stream>>>(e_out);
    prep_weights<<<(196 * 8 * 512 + 255) / 256, 256, 0, stream>>>(conv_w, w_frag);
    groupnorm_k<<<8 * 64, 256, 0, stream>>>(c, gn_w, gn_b, c_norm);
    normx_k<<<(NPIX * 64) / 256, 256, 0, stream>>>(x, x_f32, x_bf);

    for (int t = 0; t < 4; ++t) {
        conv_wmma<<<NPIX / 128, 256, SMEM, stream>>>(x_bf, w_frag, coup);
        kuramoto_k<<<NPIX / 4, 256, 0, stream>>>(x_f32, coup, c_norm, conv_b, omega,
                                                 x_bf, xs_out, e_out, t);
    }
}
